// MultiHeadSelection_49744311222808
// MI455X (gfx1250) — compile-verified, matrix operand order checked by reading, not run on hardware
//
#include <hip/hip_runtime.h>
#include <hip/hip_bf16.h>
#include <math.h>

typedef __attribute__((ext_vector_type(16))) _Float16 v16h;
typedef __attribute__((ext_vector_type(8)))  _Float16 v8h;
typedef __attribute__((ext_vector_type(8)))  float    v8f;
typedef __attribute__((ext_vector_type(4)))  float    v4f;

#define B_   16
#define L_   128
#define E_   256
#define H_   256
#define G4_  1024
#define RR_  128
#define BIO_ 64
#define NT_  3
#define R_   25
#define M_   2048   // B*L
#define OCK_ 320    // H + BIO

// ---------------------------------------------------------------- WMMA core
static __device__ __forceinline__ v8f wmma16(v16h a, v16h b, v8f c) {
  return __builtin_amdgcn_wmma_f32_16x16x32_f16(
      /*neg_a=*/false, a, /*neg_b=*/false, b,
      /*c_mod=*/(short)0, c, /*reuse_a=*/false, /*reuse_b=*/false);
}

// Load a 16x32 f16 operand tile from a row-major (rows, ld) matrix.
// A-operand: lane&15 = M row.  B-operand (weights stored (N,K) row-major):
// lane&15 = N column -> same load pattern.
static __device__ __forceinline__ v16h load_tile16(const _Float16* p, int ld,
                                                   int row0, int k0) {
  int lane = threadIdx.x & 31;
  int r  = row0 + (lane & 15);
  int kk = k0 + ((lane & 16) ? 8 : 0);
  const _Float16* base = p + (size_t)r * ld + kk;
  v8h lo = *(const v8h*)(base);
  v8h hi = *(const v8h*)(base + 16);
  v16h out;
#pragma unroll
  for (int i = 0; i < 8; ++i) { out[i] = lo[i]; out[i + 8] = hi[i]; }
  return out;
}

// ------------------------------------------------------------- small kernels
__global__ void k_zero(float* p, int n) {
  int i = blockIdx.x * blockDim.x + threadIdx.x;
  if (i < n) p[i] = 0.0f;
}

__global__ void k_cvt(_Float16* dst, const float* src, int n) {
  int i = blockIdx.x * blockDim.x + threadIdx.x;
  if (i < n) dst[i] = (_Float16)src[i];
}

__global__ void k_embed(const int* tokens, const float* wemb, _Float16* dst) {
  int i = blockIdx.x * blockDim.x + threadIdx.x;
  if (i >= M_ * E_) return;
  int m = i >> 8, e = i & 255;
  dst[i] = (_Float16)wemb[(size_t)tokens[m] * E_ + e];
}

__global__ void k_comb(const float* hf, const float* hb, float* o, int n) {
  int i = blockIdx.x * blockDim.x + threadIdx.x;
  if (i < n) o[i] = 0.5f * (hf[i] + hb[i]);
}

__global__ void k_oc(const float* o, const int* bio_gold, const float* bio_emb,
                     _Float16* oc) {
  int i = blockIdx.x * blockDim.x + threadIdx.x;
  if (i >= M_ * OCK_) return;
  int m = i / OCK_, k = i % OCK_;
  float v = (k < H_) ? o[(size_t)m * H_ + k]
                     : bio_emb[(size_t)bio_gold[m] * BIO_ + (k - H_)];
  oc[i] = (_Float16)v;
}

__global__ void k_w12(const float* src, _Float16* w1, _Float16* w2) {
  int i = blockIdx.x * blockDim.x + threadIdx.x;
  if (i >= RR_ * RR_) return;
  int n = i >> 7, k = i & 127;
  w1[i] = (_Float16)src[(size_t)n * (2 * RR_) + k];
  w2[i] = (_Float16)src[(size_t)n * (2 * RR_) + RR_ + k];
}

__global__ void k_rel(const float* src, _Float16* dst) {
  int i = blockIdx.x * blockDim.x + threadIdx.x;
  if (i >= 32 * RR_) return;
  int r = i >> 7, h = i & 127;
  dst[i] = (r < R_) ? (_Float16)src[(size_t)r * RR_ + h] : (_Float16)0.0f;
}

// -------------------------------------------------- generic WMMA GEMM
// C[M,N] = A(M,K f16 rowmajor) x W(N,K f16 rowmajor)^T (+bias)(relu?)
__global__ void k_gemm(const _Float16* __restrict__ A,
                       const _Float16* __restrict__ W,
                       const float* __restrict__ bias,
                       float* Cf, _Float16* Ch,
                       int Mt, int Nt, int K, int relu) {
  int wave = blockIdx.x * (blockDim.x >> 5) + (threadIdx.x >> 5);
  if (wave >= Mt * Nt) return;            // wave-uniform
  int mt = wave / Nt, nt = wave % Nt;
  int N = Nt << 4;
  v8f acc = {};
  for (int k0 = 0; k0 < K; k0 += 32) {
    v16h a = load_tile16(A, K, mt << 4, k0);
    v16h b = load_tile16(W, K, nt << 4, k0);
    acc = wmma16(a, b, acc);
  }
  int lane = threadIdx.x & 31;
  int n = (nt << 4) + (lane & 15);
  float bv = bias ? bias[n] : 0.0f;
#pragma unroll
  for (int r = 0; r < 8; ++r) {
    int m = (mt << 4) + r + ((lane & 16) ? 8 : 0);
    float v = acc[r] + bv;
    if (relu) v = fmaxf(v, 0.0f);
    size_t idx = (size_t)m * N + n;
    if (Cf) Cf[idx] = v;
    if (Ch) Ch[idx] = (_Float16)v;
  }
}

// ------------------------------------------------------------ LSTM recurrence
// blockIdx.x = direction (0=fwd, 1=bwd); both chains run concurrently.
// 16 waves; wave w owns H-columns [16w,16w+16) of all four gates, so the
// cell state stays in the accumulator register layout.
__global__ void k_lstm(const float* __restrict__ xg0,
                       const float* __restrict__ xg1,
                       const _Float16* __restrict__ W0,
                       const _Float16* __restrict__ W1,
                       float* __restrict__ h0, float* __restrict__ h1) {
  __shared__ __align__(16) _Float16 hsh[B_ * H_];
  int dir = blockIdx.x;
  const float*    xg  = dir ? xg1 : xg0;
  const _Float16* Whh = dir ? W1 : W0;
  float*          hdr = dir ? h1 : h0;
  int tid = threadIdx.x, wav = tid >> 5, lane = tid & 31;
  for (int i = tid; i < B_ * H_; i += blockDim.x) hsh[i] = (_Float16)0.0f;
  __syncthreads();
  v8f creg = {};
  int nlocal = lane & 15;
  int hiSel  = (lane & 16) ? 8 : 0;
  for (int step = 0; step < L_; ++step) {
    int t = dir ? (L_ - 1 - step) : step;
    v8f acc[4];
#pragma unroll
    for (int q = 0; q < 4; ++q) {
      int n0 = q * H_ + (wav << 4);
#pragma unroll
      for (int r = 0; r < 8; ++r) {
        int m = r + hiSel;
        acc[q][r] = xg[((size_t)m * L_ + t) * G4_ + n0 + nlocal];
      }
    }
    for (int k0 = 0; k0 < H_; k0 += 32) {
      v16h a = load_tile16(hsh, H_, 0, k0);     // A reused across 4 gates
#pragma unroll
      for (int q = 0; q < 4; ++q) {
        v16h b = load_tile16(Whh, H_, q * H_ + (wav << 4), k0);
        acc[q] = wmma16(a, b, acc[q]);
      }
    }
    __syncthreads();   // all reads of hsh done before rewrite
    int j = (wav << 4) + nlocal;
#pragma unroll
    for (int r = 0; r < 8; ++r) {
      int m = r + hiSel;
      float ig = 1.0f / (1.0f + __expf(-acc[0][r]));
      float fg = 1.0f / (1.0f + __expf(-acc[1][r]));
      float gg = tanhf(acc[2][r]);
      float og = 1.0f / (1.0f + __expf(-acc[3][r]));
      float cn = fg * creg[r] + ig * gg;
      float hv = og * tanhf(cn);
      creg[r] = cn;
      hsh[m * H_ + j] = (_Float16)hv;
      hdr[((size_t)m * L_ + t) * H_ + j] = hv;
    }
    __syncthreads();
  }
}

// --------------------------------------------------------------- emissions
__global__ void k_emi(const float* o, const float* emi_W, const float* emi_b,
                      float* emi) {
  int tid = blockIdx.x * blockDim.x + threadIdx.x;
  if (tid >= M_ * NT_) return;
  int m = tid / NT_, n = tid % NT_;
  const float* orow = o + (size_t)m * H_;
  const float* wrow = emi_W + (size_t)n * H_;
  float s = emi_b[n];
  for (int k = 0; k < H_; ++k) s += orow[k] * wrow[k];
  emi[tid] = s;
}

// ------------------------------------------------------------------ CRF loss
__global__ void k_crf(const int* tokens, const int* bio_gold, const float* emi,
                      const float* crf_start, const float* crf_end,
                      const float* crf_trans, float* accum) {
  __shared__ float redc[B_], redm[B_];
  int b = threadIdx.x;
  if (b < B_) {
    const int* tok = tokens + b * L_;
    const int* bg  = bio_gold + b * L_;
    const float* em = emi + (size_t)b * L_ * NT_;
    float num = crf_start[bg[0]] + em[bg[0]];
    int cnt = (tok[0] != 0) ? 1 : 0;
    for (int t = 1; t < L_; ++t) {
      if (tok[t] != 0) {
        num += crf_trans[bg[t - 1] * NT_ + bg[t]] + em[t * NT_ + bg[t]];
        cnt++;
      }
    }
    int se = cnt - 1; if (se < 0) se = 0;
    num += crf_end[bg[se]];
    float score[NT_];
    for (int k = 0; k < NT_; ++k) score[k] = crf_start[k] + em[k];
    for (int t = 1; t < L_; ++t) {
      if (tok[t] != 0) {
        float nxt[NT_];
        for (int k = 0; k < NT_; ++k) {
          float mx = -1e30f;
          for (int j = 0; j < NT_; ++j)
            mx = fmaxf(mx, score[j] + crf_trans[j * NT_ + k]);
          float s = 0.0f;
          for (int j = 0; j < NT_; ++j)
            s += __expf(score[j] + crf_trans[j * NT_ + k] - mx);
          nxt[k] = mx + __logf(s) + em[t * NT_ + k];
        }
        for (int k = 0; k < NT_; ++k) score[k] = nxt[k];
      }
    }
    float mx = -1e30f;
    for (int k = 0; k < NT_; ++k) mx = fmaxf(mx, score[k] + crf_end[k]);
    float s = 0.0f;
    for (int k = 0; k < NT_; ++k) s += __expf(score[k] + crf_end[k] - mx);
    redc[b] = num - (mx + __logf(s));
    float mc = 0.0f;
    for (int t = 0; t < L_; ++t) mc += (tok[t] != 0) ? 1.0f : 0.0f;
    redm[b] = mc;
  }
  __syncthreads();
  if (threadIdx.x == 0) {
    float sc = 0.0f, sm = 0.0f;
    for (int q = 0; q < B_; ++q) { sc += redc[q]; sm += redm[q]; }
    accum[1] = sc;  // sum_b (num - den)
    accum[2] = sm;  // maskf.sum()
  }
}

// ----------------------------------------- selection bilinear + BCE reduction
// block = (b,i).  The 12.8KB gold tile is copied global->LDS with
// GLOBAL_LOAD_ASYNC_TO_LDS_B128 (ASYNCcnt) at block entry so the HBM stream
// overlaps the logits WMMA phase; BCE then reads it from LDS.
__global__ void k_sel(const float* __restrict__ abuf,
                      const float* __restrict__ cbuf,
                      const float* __restrict__ uvb,
                      const _Float16* __restrict__ relW,
                      const float* __restrict__ selg,
                      const int* __restrict__ tokens, float* accum) {
  __shared__ __align__(16) float cvec[RR_];
  __shared__ float maskj[L_];
  __shared__ float llog[L_ * 32];
  __shared__ float sred[512];
  __shared__ __align__(16) float yld[R_ * L_];   // async-landed gold tile
  int bi = blockIdx.x;
  int b = bi >> 7, i = bi & 127;
  if (tokens[(size_t)b * L_ + i] == 0) return;  // block-uniform skip (mask_i)
  int tid = threadIdx.x;
  const float* ybase = selg + (size_t)bi * (R_ * L_);
  // 3200 floats = 800 x 16B chunks: issue async global->LDS copies now.
  for (int c = tid; c < (R_ * L_) / 4; c += blockDim.x) {
    unsigned long long ga = (unsigned long long)(ybase + c * 4);
    unsigned lds = (unsigned)(size_t)(yld + c * 4);   // generic->LDS addr[31:0]
    asm volatile("global_load_async_to_lds_b128 %0, %1, off"
                 :: "v"(lds), "v"(ga) : "memory");
  }
  if (tid < RR_) cvec[tid] = cbuf[(size_t)bi * RR_ + tid] + uvb[tid];
  if (tid < L_) maskj[tid] = (tokens[(size_t)b * L_ + tid] != 0) ? 1.0f : 0.0f;
  __syncthreads();
  int wav = tid >> 5, lane = tid & 31;
  int jt = wav >> 1, ntile = wav & 1;
  int jrow = (jt << 4) + (lane & 15);
  int kof  = (lane & 16) ? 8 : 0;
  const v4f* arow4 = (const v4f*)(abuf + (size_t)(b * L_ + jrow) * RR_);
  const v4f* cvv4  = (const v4f*)cvec;
  v8f acc = {};
  for (int h0 = 0; h0 < RR_; h0 += 32) {
    int q0 = (h0 + kof) >> 2;          // float4 index
    v4f a0 = arow4[q0],     a1 = arow4[q0 + 1];
    v4f a2 = arow4[q0 + 4], a3 = arow4[q0 + 5];   // +16 floats
    v4f c0 = cvv4[q0],      c1 = cvv4[q0 + 1];
    v4f c2 = cvv4[q0 + 4],  c3 = cvv4[q0 + 5];
    v16h av;
#pragma unroll
    for (int q = 0; q < 4; ++q) {
      av[q]      = (_Float16)fmaxf(a0[q] + c0[q], 0.0f);
      av[q + 4]  = (_Float16)fmaxf(a1[q] + c1[q], 0.0f);
      av[q + 8]  = (_Float16)fmaxf(a2[q] + c2[q], 0.0f);
      av[q + 12] = (_Float16)fmaxf(a3[q] + c3[q], 0.0f);
    }
    v16h bv = load_tile16(relW, RR_, ntile << 4, h0);
    acc = wmma16(av, bv, acc);
  }
  int n = (ntile << 4) + (lane & 15);
#pragma unroll
  for (int r = 0; r < 8; ++r) {
    int j = (jt << 4) + r + ((lane & 16) ? 8 : 0);
    llog[j * 32 + n] = acc[r];
  }
  // wait for this wave's async tile chunks, then sync all waves
  asm volatile("s_wait_asynccnt 0x0" ::: "memory");
  __syncthreads();
  float local = 0.0f;
  for (int idx = tid; idx < R_ * L_; idx += blockDim.x) {
    int r = idx >> 7, j = idx & 127;
    float x = llog[j * 32 + r];
    float y = yld[idx];
    float bce = fmaxf(x, 0.0f) - x * y + log1pf(__expf(-fabsf(x)));
    local += bce * maskj[j];
  }
  sred[tid] = local;
  __syncthreads();
  for (int s = 256; s > 0; s >>= 1) {
    if (tid < s) sred[tid] += sred[tid + s];
    __syncthreads();
  }
  if (tid == 0) atomicAdd(accum, sred[0]);
}

__global__ void k_final(const float* accum, float* out) {
  out[0] = -(accum[1] / (float)B_) + accum[0] / accum[2];
}

// ------------------------------------------------------------------ launcher
extern "C" void kernel_launch(void* const* d_in, const int* in_sizes, int n_in,
                              void* d_out, int out_size, void* d_ws,
                              size_t ws_size, hipStream_t stream) {
  const int*   tokens    = (const int*)d_in[0];
  const int*   bio_gold  = (const int*)d_in[1];
  const float* sel_gold  = (const float*)d_in[2];
  const float* word_emb  = (const float*)d_in[3];
  const float* W_ih_f    = (const float*)d_in[4];
  const float* W_hh_f    = (const float*)d_in[5];
  const float* b_f       = (const float*)d_in[6];
  const float* W_ih_b    = (const float*)d_in[7];
  const float* W_hh_b    = (const float*)d_in[8];
  const float* b_b       = (const float*)d_in[9];
  const float* emi_W     = (const float*)d_in[10];
  const float* emi_b     = (const float*)d_in[11];
  const float* bio_emb   = (const float*)d_in[12];
  const float* sel_u_W   = (const float*)d_in[13];
  const float* sel_u_b   = (const float*)d_in[14];
  const float* sel_v_W   = (const float*)d_in[15];
  const float* sel_v_b   = (const float*)d_in[16];
  const float* sel_uv_W  = (const float*)d_in[17];
  const float* sel_uv_b  = (const float*)d_in[18];
  const float* rel_emb   = (const float*)d_in[19];
  const float* crf_start = (const float*)d_in[20];
  const float* crf_end   = (const float*)d_in[21];
  const float* crf_trans = (const float*)d_in[22];

  char* ws = (char*)d_ws;
  size_t off = 0;
  auto alloc = [&](size_t bytes) {
    size_t o = off; off += (bytes + 255) & ~(size_t)255; return o;
  };
  float*     accum  = (float*)(ws + alloc(16 * 4));
  _Float16*  emb16  = (_Float16*)(ws + alloc((size_t)M_ * E_ * 2));
  _Float16*  wihf16 = (_Float16*)(ws + alloc((size_t)G4_ * E_ * 2));
  _Float16*  wihb16 = (_Float16*)(ws + alloc((size_t)G4_ * E_ * 2));
  _Float16*  whhf16 = (_Float16*)(ws + alloc((size_t)G4_ * H_ * 2));
  _Float16*  whhb16 = (_Float16*)(ws + alloc((size_t)G4_ * H_ * 2));
  float*     xgf    = (float*)(ws + alloc((size_t)M_ * G4_ * 4));
  float*     xgb    = (float*)(ws + alloc((size_t)M_ * G4_ * 4));
  float*     hf     = (float*)(ws + alloc((size_t)M_ * H_ * 4));
  float*     hb     = (float*)(ws + alloc((size_t)M_ * H_ * 4));
  float*     obuf   = (float*)(ws + alloc((size_t)M_ * H_ * 4));
  float*     emib   = (float*)(ws + alloc((size_t)M_ * NT_ * 4));
  _Float16*  oc16   = (_Float16*)(ws + alloc((size_t)M_ * OCK_ * 2));
  _Float16*  su16   = (_Float16*)(ws + alloc((size_t)RR_ * OCK_ * 2));
  _Float16*  sv16   = (_Float16*)(ws + alloc((size_t)RR_ * OCK_ * 2));
  _Float16*  w1h    = (_Float16*)(ws + alloc((size_t)RR_ * RR_ * 2));
  _Float16*  w2h    = (_Float16*)(ws + alloc((size_t)RR_ * RR_ * 2));
  _Float16*  relw   = (_Float16*)(ws + alloc((size_t)32 * RR_ * 2));
  _Float16*  u16    = (_Float16*)(ws + alloc((size_t)M_ * RR_ * 2));
  _Float16*  v16    = (_Float16*)(ws + alloc((size_t)M_ * RR_ * 2));
  float*     abuf   = (float*)(ws + alloc((size_t)M_ * RR_ * 4));
  float*     cbuf   = (float*)(ws + alloc((size_t)M_ * RR_ * 4));
  (void)ws_size; (void)in_sizes; (void)n_in; (void)out_size;

  auto cdiv = [](int a, int b) { return (a + b - 1) / b; };

  k_zero<<<1, 64, 0, stream>>>(accum, 16);
  k_cvt<<<cdiv(G4_ * E_, 256), 256, 0, stream>>>(wihf16, W_ih_f, G4_ * E_);
  k_cvt<<<cdiv(G4_ * E_, 256), 256, 0, stream>>>(wihb16, W_ih_b, G4_ * E_);
  k_cvt<<<cdiv(G4_ * H_, 256), 256, 0, stream>>>(whhf16, W_hh_f, G4_ * H_);
  k_cvt<<<cdiv(G4_ * H_, 256), 256, 0, stream>>>(whhb16, W_hh_b, G4_ * H_);
  k_cvt<<<cdiv(RR_ * OCK_, 256), 256, 0, stream>>>(su16, sel_u_W, RR_ * OCK_);
  k_cvt<<<cdiv(RR_ * OCK_, 256), 256, 0, stream>>>(sv16, sel_v_W, RR_ * OCK_);
  k_w12<<<cdiv(RR_ * RR_, 256), 256, 0, stream>>>(sel_uv_W, w1h, w2h);
  k_rel<<<cdiv(32 * RR_, 256), 256, 0, stream>>>(rel_emb, relw);
  k_embed<<<cdiv(M_ * E_, 256), 256, 0, stream>>>(tokens, word_emb, emb16);

  // xg = emb @ W_ih^T + b   (M=2048, N=1024, K=256) -> 8192 tiles
  k_gemm<<<(128 * 64) / 4, 128, 0, stream>>>(emb16, wihf16, b_f, xgf, nullptr,
                                             128, 64, E_, 0);
  k_gemm<<<(128 * 64) / 4, 128, 0, stream>>>(emb16, wihb16, b_b, xgb, nullptr,
                                             128, 64, E_, 0);
  // both sequential recurrences concurrently (blockIdx.x = direction)
  k_lstm<<<2, 512, 0, stream>>>(xgf, xgb, whhf16, whhb16, hf, hb);

  k_comb<<<cdiv(M_ * H_, 256), 256, 0, stream>>>(hf, hb, obuf, M_ * H_);
  k_emi<<<cdiv(M_ * NT_, 256), 256, 0, stream>>>(obuf, emi_W, emi_b, emib);
  k_crf<<<1, 32, 0, stream>>>(tokens, bio_gold, emib, crf_start, crf_end,
                              crf_trans, accum);

  k_oc<<<cdiv(M_ * OCK_, 256), 256, 0, stream>>>(obuf, bio_gold, bio_emb, oc16);
  // u,v : (2048x320)x(320x128), relu, f16 out
  k_gemm<<<(128 * 8) / 4, 128, 0, stream>>>(oc16, su16, sel_u_b, nullptr, u16,
                                            128, 8, OCK_, 1);
  k_gemm<<<(128 * 8) / 4, 128, 0, stream>>>(oc16, sv16, sel_v_b, nullptr, v16,
                                            128, 8, OCK_, 1);
  // a = u @ W1^T, c = v @ W2^T : (2048x128)x(128x128), f32 out
  k_gemm<<<(128 * 8) / 4, 128, 0, stream>>>(u16, w1h, nullptr, abuf, nullptr,
                                            128, 8, RR_, 0);
  k_gemm<<<(128 * 8) / 4, 128, 0, stream>>>(v16, w2h, nullptr, cbuf, nullptr,
                                            128, 8, RR_, 0);
  // bilinear logits + BCE, one block per (b,i)
  k_sel<<<M_, 512, 0, stream>>>(abuf, cbuf, sel_uv_b, relw, sel_gold, tokens,
                                accum);
  k_final<<<1, 1, 0, stream>>>(accum, (float*)d_out);
}
